// LSTM_42777874268815
// MI455X (gfx1250) — matrix operand order repaired and checked by reading, compiled, not run on hardware
//
#include <hip/hip_runtime.h>

// ---------------------------------------------------------------------------
// 2-layer LSTM (B=64,T=512,D=256,H=512,O=256) for gfx1250 using bf16 WMMA.
// Weights/activations converted to bf16 once; all GEMMs via
// v_wmma_f32_16x16x32_bf16 with f32 accumulation. Recurrent chain runs as
// one graph-captured kernel launch per time step per layer; all weight and
// state operands are L2-resident (192 MB), so direct global_load_b128 ->
// WMMA is the optimal data path (no LDS staging required).
// ---------------------------------------------------------------------------

typedef __attribute__((ext_vector_type(16))) __bf16 v16bf;
typedef __attribute__((ext_vector_type(8)))  float  v8f;

namespace {
constexpr int Bb = 64;     // batch
constexpr int Tt = 512;    // time
constexpr int Dd = 256;    // input dim
constexpr int Hh = 512;    // hidden
constexpr int Oo = 256;    // output dim
}

// ---- WMMA operand loaders (ISA 7.12.2 layouts, 16-bit elements) -----------
// A (MxK): lane l holds row m=l&15; vec elems 0-7 -> K=k0+8*half+e,
//          elems 8-15 -> K=k0+16+8*half+(e-8).
__device__ __forceinline__ v16bf loadA(const __bf16* __restrict__ row, int k0, int half) {
    union { v16bf v; uint4 u[2]; } t;
    t.u[0] = *(const uint4*)(row + k0 + half * 8);
    t.u[1] = *(const uint4*)(row + k0 + 16 + half * 8);
    return t.v;
}
// B (KxN), stored as [N][K] row-major (weight rows): lane l holds col n=l&15;
// vec elems 0-7 -> K=k0+8*half+e, elems 8-15 -> K=k0+16+8*half+(e-8).
__device__ __forceinline__ v16bf loadB(const __bf16* __restrict__ row, int k0, int half) {
    union { v16bf v; uint4 u[2]; } t;
    t.u[0] = *(const uint4*)(row + k0 + half * 8);
    t.u[1] = *(const uint4*)(row + k0 + 16 + half * 8);
    return t.v;
}

// Fast gate nonlinearities on the serial critical path: one __expf each.
__device__ __forceinline__ float fast_sigmoid(float x) {
    return 1.0f / (1.0f + __expf(-x));
}
__device__ __forceinline__ float fast_tanh(float x) {
    // tanh(x) = 2*sigmoid(2x) - 1
    return 2.0f / (1.0f + __expf(-2.0f * x)) - 1.0f;
}

// ---- conversion / prep kernels --------------------------------------------
__global__ void k_cvt_bf16(const float* __restrict__ s, __bf16* __restrict__ d, int n) {
    for (int i = blockIdx.x * blockDim.x + threadIdx.x; i < n; i += gridDim.x * blockDim.x)
        d[i] = (__bf16)s[i];
}

__global__ void k_bias_sum(const float* __restrict__ a, const float* __restrict__ b,
                           float* __restrict__ d, int n) {
    for (int i = blockIdx.x * blockDim.x + threadIdx.x; i < n; i += gridDim.x * blockDim.x)
        d[i] = a[i] + b[i];
}

// ---- one LSTM time step ----------------------------------------------------
// grid = (Hh/16, Bb/16), block = 32 (one wave). Each wave computes the four
// gate tiles (i,f,g,o) for one 16x16 (batch x hidden) block:
//   gates = A1[Bb,K1] * Wih^T + Hprev[Bb,Hh] * Whh^T + bias
// then applies the LSTM cell update directly on the accumulator registers.
__global__ void k_lstm_step(const __bf16* __restrict__ A1, int lda1, int K1,
                            const __bf16* __restrict__ Wih,    // [4H, K1] bf16
                            const __bf16* __restrict__ Hprev,  // [Bb, Hh] bf16 or null (t==0)
                            const __bf16* __restrict__ Whh,    // [4H, Hh] bf16
                            const float*  __restrict__ bias,   // [4H] (b_ih + b_hh)
                            float*        __restrict__ c,      // [Bb, Hh] f32, in/out
                            __bf16*       __restrict__ hbf,    // [Bb, Hh] bf16 out
                            float*        __restrict__ hf)     // f32 h_T out or null
{
    const int l    = threadIdx.x;
    const int half = l >> 4;
    const int lm   = l & 15;
    const int mt   = blockIdx.y;   // batch tile
    const int nt   = blockIdx.x;   // hidden tile

    v8f acc[4] = { {}, {}, {}, {} };

    // input contribution: A1 row per lane, 4 weight rows (one per gate)
    const __bf16* arow = A1 + (size_t)(mt * 16 + lm) * lda1;
    const __bf16* wg[4];
#pragma unroll
    for (int g = 0; g < 4; ++g)
        wg[g] = Wih + (size_t)(g * Hh + nt * 16 + lm) * K1;

    for (int k0 = 0; k0 < K1; k0 += 32) {
        v16bf a = loadA(arow, k0, half);
#pragma unroll
        for (int g = 0; g < 4; ++g)
            acc[g] = __builtin_amdgcn_wmma_f32_16x16x32_bf16(
                false, a, false, loadB(wg[g], k0, half), (short)0, acc[g], false, false);
    }

    // recurrent contribution (skipped at t==0 where h == 0); uniform branch,
    // EXEC stays all-ones as WMMA requires.
    if (Hprev != nullptr) {
        const __bf16* hrow = Hprev + (size_t)(mt * 16 + lm) * Hh;
        const __bf16* vg[4];
#pragma unroll
        for (int g = 0; g < 4; ++g)
            vg[g] = Whh + (size_t)(g * Hh + nt * 16 + lm) * Hh;

        for (int k0 = 0; k0 < Hh; k0 += 32) {
            v16bf a = loadA(hrow, k0, half);
#pragma unroll
            for (int g = 0; g < 4; ++g)
                acc[g] = __builtin_amdgcn_wmma_f32_16x16x32_bf16(
                    false, a, false, loadB(vg[g], k0, half), (short)0, acc[g], false, false);
        }
    }

    // cell update on the C/D layout: elem r -> m=r+8*half, n=lm
    const int n  = nt * 16 + lm;
    const float b0 = bias[0 * Hh + n];
    const float b1 = bias[1 * Hh + n];
    const float b2 = bias[2 * Hh + n];
    const float b3 = bias[3 * Hh + n];
    const bool first = (Hprev == nullptr);

#pragma unroll
    for (int r = 0; r < 8; ++r) {
        const int    m   = mt * 16 + r + 8 * half;
        const size_t idx = (size_t)m * Hh + n;
        const float iv = fast_sigmoid(acc[0][r] + b0);
        const float fv = fast_sigmoid(acc[1][r] + b1);
        const float gv = fast_tanh   (acc[2][r] + b2);
        const float ov = fast_sigmoid(acc[3][r] + b3);
        const float cp = first ? 0.0f : c[idx];
        const float cn = fv * cp + iv * gv;
        const float hn = ov * fast_tanh(cn);
        c[idx]   = cn;
        hbf[idx] = (__bf16)hn;
        if (hf != nullptr) hf[idx] = hn;   // only at t == T-1 (-> hidden output)
    }
}

// ---- output head: out[b,t,:] = h2[b,t,:] @ Wout^T + b_out ------------------
// Hseq rows are indexed m = t*Bb + b. grid = (Oo/64, Tt*Bb/16), block = 32.
__global__ void k_out_gemm(const __bf16* __restrict__ Hseq,  // [Tt*Bb, Hh] bf16
                           const __bf16* __restrict__ Wo,    // [Oo, Hh] bf16
                           const float*  __restrict__ bo,    // [Oo]
                           float*        __restrict__ out)   // [Bb, Tt, Oo]
{
    const int l    = threadIdx.x;
    const int half = l >> 4;
    const int lm   = l & 15;
    const int mt   = blockIdx.y;
    const int ng   = blockIdx.x;

    v8f acc[4] = { {}, {}, {}, {} };
    const __bf16* arow = Hseq + (size_t)(mt * 16 + lm) * Hh;
    const __bf16* w[4];
#pragma unroll
    for (int g = 0; g < 4; ++g)
        w[g] = Wo + (size_t)(ng * 64 + g * 16 + lm) * Hh;

    for (int k0 = 0; k0 < Hh; k0 += 32) {
        v16bf a = loadA(arow, k0, half);
#pragma unroll
        for (int g = 0; g < 4; ++g)
            acc[g] = __builtin_amdgcn_wmma_f32_16x16x32_bf16(
                false, a, false, loadB(w[g], k0, half), (short)0, acc[g], false, false);
    }

#pragma unroll
    for (int g = 0; g < 4; ++g) {
        const int   n  = ng * 64 + g * 16 + lm;
        const float bb = bo[n];
#pragma unroll
        for (int r = 0; r < 8; ++r) {
            const int m = mt * 16 + r + 8 * half;   // m = t*Bb + b
            const int b = m & (Bb - 1);
            const int t = m >> 6;                   // Bb == 64
            out[((size_t)b * Tt + t) * Oo + n] = acc[g][r] + bb;
        }
    }
}

// ---------------------------------------------------------------------------
extern "C" void kernel_launch(void* const* d_in, const int* in_sizes, int n_in,
                              void* d_out, int out_size, void* d_ws, size_t ws_size,
                              hipStream_t stream) {
    (void)in_sizes; (void)n_in; (void)out_size; (void)ws_size;

    const float* x    = (const float*)d_in[0];
    const float* Wih0 = (const float*)d_in[1];
    const float* Whh0 = (const float*)d_in[2];
    const float* bih0 = (const float*)d_in[3];
    const float* bhh0 = (const float*)d_in[4];
    const float* Wih1 = (const float*)d_in[5];
    const float* Whh1 = (const float*)d_in[6];
    const float* bih1 = (const float*)d_in[7];
    const float* bhh1 = (const float*)d_in[8];
    const float* Wout = (const float*)d_in[9];
    const float* bout = (const float*)d_in[10];

    // workspace carve-up (256B aligned); total ~92 MB
    char*  ws  = (char*)d_ws;
    size_t off = 0;
    auto alloc = [&](size_t bytes) -> char* {
        char* p = ws + off;
        off += (bytes + 255) & ~(size_t)255;
        return p;
    };
    __bf16* wih0b = (__bf16*)alloc((size_t)4 * Hh * Dd * 2);
    __bf16* whh0b = (__bf16*)alloc((size_t)4 * Hh * Hh * 2);
    __bf16* wih1b = (__bf16*)alloc((size_t)4 * Hh * Hh * 2);
    __bf16* whh1b = (__bf16*)alloc((size_t)4 * Hh * Hh * 2);
    __bf16* woutb = (__bf16*)alloc((size_t)Oo * Hh * 2);
    float*  bias0 = (float*)alloc((size_t)4 * Hh * 4);
    float*  bias1 = (float*)alloc((size_t)4 * Hh * 4);
    __bf16* xb    = (__bf16*)alloc((size_t)Bb * Tt * Dd * 2);
    __bf16* h0all = (__bf16*)alloc((size_t)Tt * Bb * Hh * 2);  // [t][b][h]
    __bf16* h2all = (__bf16*)alloc((size_t)Tt * Bb * Hh * 2);  // [t][b][h]
    float*  c0    = (float*)alloc((size_t)Bb * Hh * 4);
    float*  c1    = (float*)alloc((size_t)Bb * Hh * 4);

    float* outp    = (float*)d_out;                       // [Bb,Tt,Oo]
    float* hidden0 = outp + (size_t)Bb * Tt * Oo;         // [Bb,Hh]
    float* hidden1 = hidden0 + (size_t)Bb * Hh;           // [Bb,Hh]

    auto cvt = [&](const float* s, __bf16* d, int n) {
        int blocks = (n + 255) / 256;
        if (blocks > 16384) blocks = 16384;
        k_cvt_bf16<<<blocks, 256, 0, stream>>>(s, d, n);
    };
    cvt(Wih0, wih0b, 4 * Hh * Dd);
    cvt(Whh0, whh0b, 4 * Hh * Hh);
    cvt(Wih1, wih1b, 4 * Hh * Hh);
    cvt(Whh1, whh1b, 4 * Hh * Hh);
    cvt(Wout, woutb, Oo * Hh);
    cvt(x,    xb,    Bb * Tt * Dd);
    k_bias_sum<<<8, 256, 0, stream>>>(bih0, bhh0, bias0, 4 * Hh);
    k_bias_sum<<<8, 256, 0, stream>>>(bih1, bhh1, bias1, 4 * Hh);

    const dim3 sgrid(Hh / 16, Bb / 16);   // (32, 4)

    // layer 0: A1 rows stride over batch in x[B][T][D]
    for (int t = 0; t < Tt; ++t) {
        const __bf16* hprev = (t == 0) ? nullptr : (h0all + (size_t)(t - 1) * Bb * Hh);
        float* hT = (t == Tt - 1) ? hidden0 : nullptr;
        k_lstm_step<<<sgrid, 32, 0, stream>>>(
            xb + (size_t)t * Dd, Tt * Dd, Dd,
            wih0b, hprev, whh0b, bias0, c0,
            h0all + (size_t)t * Bb * Hh, hT);
    }
    // layer 1: A1 = h0all[t] ([b][h] contiguous)
    for (int t = 0; t < Tt; ++t) {
        const __bf16* hprev = (t == 0) ? nullptr : (h2all + (size_t)(t - 1) * Bb * Hh);
        float* hT = (t == Tt - 1) ? hidden1 : nullptr;
        k_lstm_step<<<sgrid, 32, 0, stream>>>(
            h0all + (size_t)t * Bb * Hh, Hh, Hh,
            wih1b, hprev, whh1b, bias1, c1,
            h2all + (size_t)t * Bb * Hh, hT);
    }

    // output head
    const dim3 ogrid(Oo / 64, (Tt * Bb) / 16);  // (4, 2048)
    k_out_gemm<<<ogrid, 32, 0, stream>>>(h2all, woutb, bout, outp);
}
